// SparseResUQueryNet_21517786153148
// MI455X (gfx1250) — matrix operand order repaired and checked by reading, not compiled, not run on hardware
//
#include <hip/hip_runtime.h>
#include <hip/hip_bf16.h>

// ---------------------------------------------------------------------------
// CDNA5 / gfx1250 sparse ResU query pipeline.
//   Stage 1: hash-gather sparse conv 1->32 over history voxels (VALU FMA)
//   Stage 2: segment-max pooling via monotonic-mapped atomicMax
//   Stage 2b: finalize pooled table to bf16 rows (+ zero sentinel row 0)
//   Stage 3: 27-tap gather-GEMM 32->32 at query points:
//            TDM gather (tensor_load_to_lds, 32-bit row indices) -> LDS,
//            ds_load_b128 A operands, v_wmma_f32_16x16x32_bf16, double-buffered
//            on TENSORcnt.
// ---------------------------------------------------------------------------

typedef __attribute__((ext_vector_type(16))) __bf16 v16bf;
typedef __attribute__((ext_vector_type(8)))  float  v8f;
typedef __attribute__((ext_vector_type(4)))  unsigned u32x4;
typedef __attribute__((ext_vector_type(8)))  int      i32x8;
typedef __attribute__((ext_vector_type(4)))  int      i32x4;

union U16BF { unsigned u[8]; v16bf v; };

#define HCAP   (1u << 21)
#define HMASK  (HCAP - 1u)

__device__ __forceinline__ unsigned hmix(unsigned x) {
    x ^= x >> 16; x *= 0x7feb352du;
    x ^= x >> 15; x *= 0x846ca68bu;
    x ^= x >> 16; return x;
}

__device__ __forceinline__ int pack4i(int a, int b, int c, int d) {
    return ((a * 256 + b) * 256 + c) * 256 + d;   // S = 256
}

__device__ __forceinline__ int dlt(int off) {     // DELTAS[off], dx outer
    int dx = off / 9 - 1, dy = (off / 3) % 3 - 1, dz = off % 3 - 1;
    return dx * 65536 + dy * 256 + dz;
}

__device__ __forceinline__ void hinsert(int* keys, int* vals, int key, int val) {
    unsigned h = hmix((unsigned)key) & HMASK;
    for (unsigned p = 0; p < HCAP; ++p) {
        int old = atomicCAS(&keys[h], -1, key);
        if (old == -1) { vals[h] = val; return; }
        if (old == key) return;
        h = (h + 1) & HMASK;
    }
}

__device__ __forceinline__ int hlookup(const int* __restrict__ keys,
                                       const int* __restrict__ vals, int key) {
    unsigned h = hmix((unsigned)key) & HMASK;
    for (unsigned p = 0; p < HCAP; ++p) {
        int k = keys[h];
        if (k == key) return vals[h];
        if (k == -1)  return -1;
        h = (h + 1) & HMASK;
    }
    return -1;
}

// monotonic float<->uint mapping so unsigned atomicMax == float max
__device__ __forceinline__ unsigned encmax(float v) {
    unsigned b = __float_as_uint(v);
    return (b & 0x80000000u) ? ~b : (b | 0x80000000u);
}
__device__ __forceinline__ float decmax(unsigned m) {
    unsigned b = (m & 0x80000000u) ? (m & 0x7fffffffu) : ~m;
    return __uint_as_float(b);
}

__device__ __forceinline__ unsigned bf16u(float f) {       // f32 -> bf16 (RNE-ish)
    unsigned u = __float_as_uint(f);
    return (u + 0x7fffu + ((u >> 16) & 1u)) >> 16;
}
__device__ __forceinline__ unsigned pk2bf(float lo, float hi) {
    return bf16u(lo) | (bf16u(hi) << 16);
}

// TDM gather: 16 per-row slots (lanes 0..15, -1 = miss) -> two 8-row gather
// descriptors (32-bit indices) into LDS at lbase / lbase+512. Row 0 of the
// bf16 table is a zero sentinel, so "slot+1" maps misses to zeros.
// NOTE: this toolchain's builtin is the 6-arg form:
//   (u32x4 g0, i32x8 g1, i32x4 g2, i32x4 g3, i32x8 g4, i32 cpol)
__device__ __forceinline__ void tdm_gather16(int sl, unsigned lbase,
                                             u32x4 g0, i32x8 g1) {
    int rows[16];
    #pragma unroll
    for (int j = 0; j < 16; ++j) rows[j] = __builtin_amdgcn_readlane(sl, j) + 1;
    i32x8 g4 = {0, 0, 0, 0, 0, 0, 0, 0};
    g0.y = lbase;
    i32x4 g2 = {rows[0],  rows[1],  rows[2],  rows[3]};
    i32x4 g3 = {rows[4],  rows[5],  rows[6],  rows[7]};
    __builtin_amdgcn_tensor_load_to_lds(g0, g1, g2, g3, g4, 0);
    g0.y = lbase + 512u;
    i32x4 h2 = {rows[8],  rows[9],  rows[10], rows[11]};
    i32x4 h3 = {rows[12], rows[13], rows[14], rows[15]};
    __builtin_amdgcn_tensor_load_to_lds(g0, g1, h2, h3, g4, 0);
}

// ---------------------------------------------------------------------------
__global__ void init_kernel(int* hAk, int* hBk, unsigned* pooled, int cap, int pn) {
    int i = blockIdx.x * 256 + threadIdx.x;
    if (i < cap) { hAk[i] = -1; hBk[i] = -1; }
    if (i < pn)  pooled[i] = 0u;                 // 0 decodes to "most negative"
}

__global__ void build_hash_kernel(const int* __restrict__ hc, const int* __restrict__ hb,
                                  int* hAk, int* hAv, int* hBk, int* hBv, int n) {
    int i = blockIdx.x * 256 + threadIdx.x;
    if (i >= n) return;
    int t = hc[i*4+0], x = hc[i*4+1], y = hc[i*4+2], z = hc[i*4+3];
    hinsert(hAk, hAv, pack4i(t, x, y, z), i);          // history key -> row
    hinsert(hBk, hBv, pack4i(hb[i], x, y, z), i);      // pooled key -> representative row
}

// Pre-swizzle W_conv[27][32][32] into the bf16 WMMA B-operand layout:
// Bpk[((off*2+tile)*32 + lane)*8 + v] packs K = 2v+(lane>=16?16:0), col = tile*16+(lane&15)
__global__ void prep_w_kernel(const float* __restrict__ Wc, unsigned* __restrict__ Bpk) {
    int id = blockIdx.x * 256 + threadIdx.x;
    if (id >= 27 * 2 * 32 * 8) return;
    int v    = id & 7;
    int lane = (id >> 3) & 31;
    int tl   = (id >> 8) & 1;
    int off  = id >> 9;
    int kb   = 2 * v + ((lane >= 16) ? 16 : 0);
    int ncol = tl * 16 + (lane & 15);
    Bpk[id]  = pk2bf(Wc[off*1024 + kb*32 + ncol], Wc[off*1024 + (kb+1)*32 + ncol]);
}

// Stage 1 + Stage 2: 27-tap conv 1->32, then scatter max-pool into pooled[]
__global__ __launch_bounds__(256)
void conv1_pool_kernel(const int* __restrict__ hc, const int* __restrict__ hb,
                       const float* __restrict__ hf, const float* __restrict__ Wbb,
                       const int* __restrict__ hAk, const int* __restrict__ hAv,
                       const int* __restrict__ hBk, const int* __restrict__ hBv,
                       unsigned* __restrict__ pooled, int n) {
    __shared__ float wsh[27 * 32];
    for (int i = threadIdx.x; i < 27 * 32; i += 256) wsh[i] = Wbb[i];
    __syncthreads();
    int i = blockIdx.x * 256 + threadIdx.x;
    if (i >= n) return;
    int t = hc[i*4+0], x = hc[i*4+1], y = hc[i*4+2], z = hc[i*4+3];
    int keyH = pack4i(t, x, y, z);
    float acc[32];
    #pragma unroll
    for (int f = 0; f < 32; ++f) acc[f] = 0.f;
    for (int off = 0; off < 27; ++off) {
        int j = hlookup(hAk, hAv, keyH + dlt(off));
        if (j >= 0) {
            float g = hf[j];
            #pragma unroll
            for (int f = 0; f < 32; ++f) acc[f] += g * wsh[off * 32 + f];
        }
    }
    int slot = hlookup(hBk, hBv, pack4i(hb[i], x, y, z));
    if (slot >= 0) {
        unsigned* prow = pooled + (size_t)slot * 32;
        #pragma unroll
        for (int f = 0; f < 32; ++f) atomicMax(prow + f, encmax(acc[f]));
    }
}

// Stage 2b: decode monotonic-mapped pooled rows -> bf16 table, row 0 = zeros.
__global__ void finalize_pool_kernel(const unsigned* __restrict__ pooled,
                                     unsigned short* __restrict__ pbf, int n) {
    int i = blockIdx.x * 256 + threadIdx.x;
    if (i >= (n + 1) * 32) return;
    int row = i >> 5, f = i & 31;
    pbf[i] = (row == 0) ? (unsigned short)0
                        : (unsigned short)bf16u(decmax(pooled[(size_t)(row - 1) * 32 + f]));
}

// Stage 3: one wave per 16-point tile. TDM gathers the 16 neighbor rows into
// LDS (double-buffered on TENSORcnt), then 2x v_wmma_f32_16x16x32_bf16 per tap.
__global__ __launch_bounds__(256)
void conv2_wmma_kernel(const int* __restrict__ qc, const float* __restrict__ pts,
                       const unsigned short* __restrict__ pbf, const unsigned* __restrict__ Bpk,
                       const int* __restrict__ hBk, const int* __restrict__ hBv,
                       float* __restrict__ out, int Npts, int nrows) {
    __shared__ unsigned lds[8][2][16][16];     // per-wave double-buffered 16x32 bf16 A tile
    const int lane = threadIdx.x & 31;
    const int wave = threadIdx.x >> 5;
    const int tile = blockIdx.x * 8 + wave;
    const int p0   = tile * 16;
    if (p0 >= Npts) return;                    // wave-uniform exit, EXEC stays all-ones

    const int  r        = lane & 15;
    const int  prow     = p0 + r;
    const bool rowvalid = prow < Npts;
    const int  pc       = rowvalid ? prow : (Npts - 1);
    const int  qkey     = pack4i(qc[pc*4+0], qc[pc*4+1], qc[pc*4+2], qc[pc*4+3]);

    // D# templates (group0 without lds_addr, group1) -- see CDNA5 ISA
    const unsigned long long tbl = (unsigned long long)(size_t)pbf;
    u32x4 g0t = { 0xC0000001u,                               // count=1, gather_mode=1, 32b idx
                  0u,                                        // lds_addr (patched per issue)
                  (unsigned)tbl,                             // global_addr[31:0]
                  (unsigned)((tbl >> 32) & 0x1FFFFFFu) | 0x80000000u }; // addr[56:32] | type=2
    i32x8 g1  = { (int)(1u << 16),                           // data_size = 2 bytes
                  (int)(32u << 16),                          // tensor_dim0 = 32 (bits 63:48)
                  (int)(((unsigned)nrows & 0xFFFFu) << 16),  // tensor_dim1 lo (bits 95:80)
                  (int)((((unsigned)nrows >> 16) & 0xFFFFu) | (32u << 16)), // dim1 hi | tile_dim0=32
                  8,                                         // tile_dim1 = 8 gather indices
                  32,                                        // tensor_dim0_stride = 32
                  0, 0 };
    const unsigned lbase = (unsigned)(size_t)&lds[wave][0][0][0];

    v8f c0 = {};
    v8f c1 = {};

    // Prologue: gather tap 0 into buffer 0
    int slot = -1;
    if (lane < 16 && rowvalid) slot = hlookup(hBk, hBv, qkey + dlt(0));
    tdm_gather16(slot, lbase, g0t, g1);

    for (int off = 0; off < 27; ++off) {
        // Overlap next tap's hash lookup with the in-flight TDM gather
        int nslot = -1;
        if (off < 26 && lane < 16 && rowvalid)
            nslot = hlookup(hBk, hBv, qkey + dlt(off + 1));

        __builtin_amdgcn_s_wait_tensorcnt(0);
        asm volatile("" ::: "memory");

        if (off < 26)
            tdm_gather16(nslot, lbase + (unsigned)((off + 1) & 1) * 1024u, g0t, g1);

        __builtin_prefetch(Bpk + ((size_t)(off * 2) * 32 + lane) * 8, 0, 1);

        // Assemble bf16 A operand, contiguous k = 16*(lane/16) + s as in Bpk (two b128 LDS loads)
        const int buf = off & 1;
        U16BF a;
        #pragma unroll
        for (int v = 0; v < 8; ++v) {
            int pk = v + ((lane >= 16) ? 8 : 0);
            a.u[v] = lds[wave][buf][r][pk];
        }

        // B operands: pre-swizzled, 32 contiguous bytes per lane
        const uint4* b0p = (const uint4*)(Bpk + ((size_t)(off * 2 + 0) * 32 + lane) * 8);
        const uint4* b1p = (const uint4*)(Bpk + ((size_t)(off * 2 + 1) * 32 + lane) * 8);
        U16BF b0, b1;
        uint4 t0 = b0p[0], t1 = b0p[1], t2 = b1p[0], t3 = b1p[1];
        b0.u[0]=t0.x; b0.u[1]=t0.y; b0.u[2]=t0.z; b0.u[3]=t0.w;
        b0.u[4]=t1.x; b0.u[5]=t1.y; b0.u[6]=t1.z; b0.u[7]=t1.w;
        b1.u[0]=t2.x; b1.u[1]=t2.y; b1.u[2]=t2.z; b1.u[3]=t2.w;
        b1.u[4]=t3.x; b1.u[5]=t3.y; b1.u[6]=t3.z; b1.u[7]=t3.w;

        c0 = __builtin_amdgcn_wmma_f32_16x16x32_bf16(false, a.v, false, b0.v,
                                                     (short)0, c0, false, false);
        c1 = __builtin_amdgcn_wmma_f32_16x16x32_bf16(false, a.v, false, b1.v,
                                                     (short)0, c1, false, false);
    }

    // Store D per 32-bit C/D layout: lane = column, VGPR i = row (+8 for hi lanes)
    #pragma unroll
    for (int i = 0; i < 8; ++i) {
        int row = i + ((lane >= 16) ? 8 : 0);
        int p   = p0 + row;
        if (p < Npts) {
            out[(size_t)p * 36 + 4      + (lane & 15)] = c0[i];
            out[(size_t)p * 36 + 4 + 16 + (lane & 15)] = c1[i];
        }
    }
    if (lane < 16 && rowvalid) {
        #pragma unroll
        for (int j = 0; j < 4; ++j)
            out[(size_t)prow * 36 + j] = pts[(size_t)prow * 5 + j];
    }
}

// ---------------------------------------------------------------------------
extern "C" void kernel_launch(void* const* d_in, const int* in_sizes, int n_in,
                              void* d_out, int out_size, void* d_ws, size_t ws_size,
                              hipStream_t stream) {
    const float* hf  = (const float*)d_in[0];   // history_features [n]
    const float* pts = (const float*)d_in[1];   // points [Npts,5]
    const float* Wbb = (const float*)d_in[2];   // [27,1,32]
    const float* Wc  = (const float*)d_in[3];   // [27,32,32]
    const int*   hc  = (const int*)d_in[4];     // history_coordinates [n,4]
    const int*   hb  = (const int*)d_in[5];     // history_batches [n]
    const int*   qc  = (const int*)d_in[6];     // curr_coords [Npts,4]
    float*       out = (float*)d_out;

    const int n    = in_sizes[0];
    const int Npts = in_sizes[1] / 5;

    char*  w = (char*)d_ws;
    size_t o = 0;
    auto alloc = [&](size_t bytes) { void* p = w + o; o = (o + bytes + 255) & ~(size_t)255; return p; };
    int*            hAk    = (int*)alloc((size_t)HCAP * 4);
    int*            hAv    = (int*)alloc((size_t)HCAP * 4);
    int*            hBk    = (int*)alloc((size_t)HCAP * 4);
    int*            hBv    = (int*)alloc((size_t)HCAP * 4);
    unsigned*       pooled = (unsigned*)alloc((size_t)n * 32 * 4);
    unsigned*       Bpk    = (unsigned*)alloc((size_t)27 * 2 * 32 * 8 * 4);
    unsigned short* pbf    = (unsigned short*)alloc((size_t)(n + 1) * 32 * 2);

    const int pn    = n * 32;
    const int initN = (pn > (int)HCAP) ? pn : (int)HCAP;
    init_kernel<<<(initN + 255) / 256, 256, 0, stream>>>(hAk, hBk, pooled, (int)HCAP, pn);
    build_hash_kernel<<<(n + 255) / 256, 256, 0, stream>>>(hc, hb, hAk, hAv, hBk, hBv, n);
    prep_w_kernel<<<(27 * 2 * 32 * 8 + 255) / 256, 256, 0, stream>>>(Wc, Bpk);
    conv1_pool_kernel<<<(n + 255) / 256, 256, 0, stream>>>(hc, hb, hf, Wbb,
                                                           hAk, hAv, hBk, hBv, pooled, n);
    finalize_pool_kernel<<<((n + 1) * 32 + 255) / 256, 256, 0, stream>>>(pooled, pbf, n);
    const int tiles = (Npts + 15) / 16;
    conv2_wmma_kernel<<<(tiles + 7) / 8, 256, 0, stream>>>(qc, pts, pbf, Bpk,
                                                           hBk, hBv, out, Npts, n + 1);
}
